// GroupedQueryAttention_60868276519716
// MI455X (gfx1250) — matrix operand order repaired and checked by reading, not compiled, not run on hardware
//
#include <hip/hip_runtime.h>

// ---------------------------------------------------------------------------
// GQA forward for MI455X (gfx1250): all GEMMs via v_wmma_f32_16x16x32_f16
// (wave32, fp32 accumulation). LDS staging in the big GEMMs uses CDNA5
// GLOBAL_LOAD_ASYNC_TO_LDS_B128 (ASYNCcnt) with double buffering.
// ---------------------------------------------------------------------------

typedef __attribute__((ext_vector_type(16))) _Float16 v16h;
typedef __attribute__((ext_vector_type(8)))  _Float16 v8h;
typedef __attribute__((ext_vector_type(4)))  _Float16 v4h;
typedef __attribute__((ext_vector_type(8)))  float    v8f;

#define SEQ    2048
#define DM     2048
#define NKVW   512     // NUM_KV_HEADS * D_K
#define DK     64
#define NHEADS 32

// log(10000)/32 for RoPE inv_freq
#define ROPE_C 0.2878231366242557f

// ---------------- CDNA5 async global->LDS copy (tracked by ASYNCcnt) -------
// Generic LDS pointer low 32 bits == LDS byte offset (ISA 10.2 aperture rule).
__device__ __forceinline__ unsigned lds_off(const void* p) {
    return (unsigned)(unsigned long long)p;
}
__device__ __forceinline__ void async_b128(unsigned lds, unsigned long long ga) {
    asm volatile("global_load_async_to_lds_b128 %0, %1, off"
                 :: "v"(lds), "v"(ga) : "memory");
}
__device__ __forceinline__ void wait_async_le4() {
    asm volatile("s_wait_asynccnt 0x4" ::: "memory");
}
__device__ __forceinline__ void wait_async_0() {
    asm volatile("s_wait_asynccnt 0x0" ::: "memory");
}

// ---------------- WMMA fragment helpers (CDNA5 16-bit layouts) -------------
// A 16x32 f16: lane(0-15)=row, hi=lane>>4 selects K halves {hi*8..+8, 16+hi*8..+8}
__device__ __forceinline__ v16h frag_a(const _Float16* base, int ld, int lane) {
    int r = lane & 15, hi = lane >> 4;
    const _Float16* p = base + r * ld + hi * 8;
    v8h a0 = *(const v8h*)(p);
    v8h a1 = *(const v8h*)(p + 16);
    return __builtin_shufflevector(a0, a1, 0,1,2,3,4,5,6,7,8,9,10,11,12,13,14,15);
}
// B 32x16 f16 (stored as Bt rows = output cols): hi selects K halves {hi*8..+8, 16+hi*8..+8}
__device__ __forceinline__ v16h frag_b(const _Float16* base, int ld, int lane) {
    int r = lane & 15, hi = lane >> 4;
    const _Float16* p = base + r * ld + hi * 8;
    v8h b0 = *(const v8h*)(p);
    v8h b1 = *(const v8h*)(p + 16);
    return __builtin_shufflevector(b0, b1, 0,1,2,3,4,5,6,7,8,9,10,11,12,13,14,15);
}
__device__ __forceinline__ v8f wmma32(v16h a, v16h b, v8f c) {
    return __builtin_amdgcn_wmma_f32_16x16x32_f16(false, a, false, b, (short)0, c,
                                                  false, false);
}

// ---------------- fp32 -> f16 straight convert -----------------------------
__global__ __launch_bounds__(256) void cvt_kernel(const float* __restrict__ src,
                                                  _Float16* __restrict__ dst, int n) {
    int i = (blockIdx.x * 256 + threadIdx.x) * 4;
    if (i + 3 < n) {
#pragma unroll
        for (int j = 0; j < 4; ++j) dst[i + j] = (_Float16)src[i + j];
    }
}

// ---------------- fp32 [R][C] -> f16 [C][R] tiled transpose-convert --------
__global__ __launch_bounds__(256) void tcvt_kernel(const float* __restrict__ src,
                                                   _Float16* __restrict__ dst,
                                                   int R, int C) {
    __shared__ float t[32][33];
    int c0 = blockIdx.x * 32, r0 = blockIdx.y * 32;
    int tx = threadIdx.x & 31, ty = threadIdx.x >> 5;
#pragma unroll
    for (int k = 0; k < 4; ++k)
        t[ty + 8 * k][tx] = src[(size_t)(r0 + ty + 8 * k) * C + c0 + tx];
    __syncthreads();
#pragma unroll
    for (int k = 0; k < 4; ++k)
        dst[(size_t)(c0 + ty + 8 * k) * R + r0 + tx] = (_Float16)t[tx][ty + 8 * k];
}

// ---------------- generic 128x128 WMMA GEMM: C = A(f16,MxDM) * Bt(f16,NxDM)^T
// Both operands have leading dimension DM (true for every call site).
// MODE 0: fp32 store     MODE 1: fused RoPE -> f16 store
// MODE 2: transposed f16 store (out[col][row], for building v^T)
#define SH_LD 40                      // padded LDS row (halves): 80B, 16B aligned
#define BUF_H (128 * SH_LD)           // halves per buffer
template <int MODE>
__global__ __launch_bounds__(256) void gemm_proj(const _Float16* __restrict__ A,
                                                 const _Float16* __restrict__ Bt,
                                                 void* __restrict__ out,
                                                 int K, int ldout) {
    __shared__ _Float16 As[2 * BUF_H];
    __shared__ _Float16 Bs[2 * BUF_H];
    const int tid  = threadIdx.x;
    const int lane = tid & 31, wave = tid >> 5;
    const int waveM = wave >> 1, waveN = wave & 1;          // 4 x 2 waves
    const int tileM = blockIdx.y * 128, tileN = blockIdx.x * 128;

    // per-thread staging chunks: rows r0 and r0+64, 8 halves at column kc
    const int r0 = tid >> 2;
    const int kc = (tid & 3) * 8;
    unsigned long long gA0 =
        (unsigned long long)(A + (size_t)(tileM + r0) * DM + kc);
    unsigned long long gA1 = gA0 + (unsigned long long)64 * DM * 2;
    unsigned long long gB0 =
        (unsigned long long)(Bt + (size_t)(tileN + r0) * DM + kc);
    unsigned long long gB1 = gB0 + (unsigned long long)64 * DM * 2;
    const unsigned lA0 = lds_off(&As[r0 * SH_LD + kc]);
    const unsigned lA1 = lA0 + 64 * SH_LD * 2;
    const unsigned lB0 = lds_off(&Bs[r0 * SH_LD + kc]);
    const unsigned lB1 = lB0 + 64 * SH_LD * 2;
    const unsigned BUFB = BUF_H * 2;                        // bytes per buffer

    // prologue: stage buffer 0 with k=0
    async_b128(lA0, gA0); async_b128(lA1, gA1);
    async_b128(lB0, gB0); async_b128(lB1, gB1);

    v8f acc[2][4] = {};
    for (int k0 = 0; k0 < K; k0 += 32) {
        const int cur = (k0 >> 5) & 1;
        if (k0 + 32 < K) {                                  // prefetch next tile
            const unsigned nb  = (unsigned)(cur ^ 1) * BUFB;
            const unsigned long long adv = (unsigned long long)(k0 + 32) * 2;
            async_b128(lA0 + nb, gA0 + adv);
            async_b128(lA1 + nb, gA1 + adv);
            async_b128(lB0 + nb, gB0 + adv);
            async_b128(lB1 + nb, gB1 + adv);
            wait_async_le4();                               // older buffer landed
        } else {
            wait_async_0();
        }
        __syncthreads();                                    // cur visible to all

        const _Float16* as = &As[cur * BUF_H];
        const _Float16* bs = &Bs[cur * BUF_H];
        v16h af[2], bf[4];
#pragma unroll
        for (int mi = 0; mi < 2; ++mi)
            af[mi] = frag_a(&as[(waveM * 32 + mi * 16) * SH_LD], SH_LD, lane);
#pragma unroll
        for (int ni = 0; ni < 4; ++ni)
            bf[ni] = frag_b(&bs[(waveN * 64 + ni * 16) * SH_LD], SH_LD, lane);
#pragma unroll
        for (int mi = 0; mi < 2; ++mi)
#pragma unroll
            for (int ni = 0; ni < 4; ++ni)
                acc[mi][ni] = wmma32(af[mi], bf[ni], acc[mi][ni]);
        __syncthreads();                                    // done reading cur
    }

    const int ln = lane & 15, hi = lane >> 4;
    const int rowBase = tileM + waveM * 32;
    const int colBase = tileN + waveN * 64;                  // head-aligned (64)
    if (MODE == 0) {
        float* O = (float*)out;
#pragma unroll
        for (int mi = 0; mi < 2; ++mi)
#pragma unroll
            for (int ni = 0; ni < 4; ++ni)
#pragma unroll
                for (int g = 0; g < 8; ++g)
                    O[(size_t)(rowBase + mi * 16 + hi * 8 + g) * ldout +
                      colBase + ni * 16 + ln] = acc[mi][ni][g];
    } else if (MODE == 1) {                                  // RoPE -> f16
        _Float16* O = (_Float16*)out;
        float invf[4];
#pragma unroll
        for (int ni = 0; ni < 4; ++ni) {
            int d = ni * 16 + ln;                            // 0..63 within head
            invf[ni] = __expf(-(float)(d & 31) * ROPE_C);
        }
#pragma unroll
        for (int mi = 0; mi < 2; ++mi)
#pragma unroll
            for (int g = 0; g < 8; ++g) {
                int m = rowBase + mi * 16 + hi * 8 + g;
#pragma unroll
                for (int ni = 0; ni < 4; ++ni) {
                    float s, c;
                    __sincosf((float)m * invf[ni], &s, &c);
                    float v   = acc[mi][ni][g];
                    float rot = (ni < 2) ? -acc[mi][ni + 2][g] : acc[mi][ni - 2][g];
                    O[(size_t)m * ldout + colBase + ni * 16 + ln] =
                        (_Float16)(v * c + rot * s);
                }
            }
    } else {                                                 // transposed f16
        _Float16* O = (_Float16*)out;
#pragma unroll
        for (int mi = 0; mi < 2; ++mi)
#pragma unroll
            for (int ni = 0; ni < 4; ++ni)
#pragma unroll
                for (int g = 0; g < 8; ++g)
                    O[(size_t)(colBase + ni * 16 + ln) * ldout +
                      rowBase + mi * 16 + hi * 8 + g] = (_Float16)acc[mi][ni][g];
    }
}

// ---------------- per-head scores: S = scale * q_h @ k_kv^T (K = 64) -------
#define SC_LD 72
__global__ __launch_bounds__(256) void scores_kernel(const _Float16* __restrict__ qh,
                                                     const _Float16* __restrict__ kh,
                                                     float* __restrict__ attn) {
    __shared__ _Float16 As[128 * SC_LD];
    __shared__ _Float16 Bs[128 * SC_LD];
    const int tid = threadIdx.x, lane = tid & 31, wave = tid >> 5;
    const int waveM = wave >> 1, waveN = wave & 1;
    const int tileM = blockIdx.y * 128, tileN = blockIdx.x * 128;
    const int h = blockIdx.z, kvh = h >> 2;

    // async-stage both 128x64 tiles: 8 B128 per thread
    {
        const int rr = tid >> 3;                 // 0..31
        const int kk = (tid & 7) * 8;            // halves
        unsigned long long gA =
            (unsigned long long)(qh + h * DK + (size_t)(tileM + rr) * DM + kk);
        unsigned long long gB =
            (unsigned long long)(kh + kvh * DK + (size_t)(tileN + rr) * NKVW + kk);
        unsigned lA = lds_off(&As[rr * SC_LD + kk]);
        unsigned lB = lds_off(&Bs[rr * SC_LD + kk]);
#pragma unroll
        for (int j = 0; j < 4; ++j) {
            async_b128(lA, gA);
            async_b128(lB, gB);
            gA += (unsigned long long)32 * DM * 2;
            gB += (unsigned long long)32 * NKVW * 2;
            lA += 32 * SC_LD * 2;
            lB += 32 * SC_LD * 2;
        }
        wait_async_0();
    }
    __syncthreads();

    v8f acc[2][4] = {};
#pragma unroll
    for (int ks = 0; ks < 64; ks += 32) {
        v16h af[2], bf[4];
#pragma unroll
        for (int mi = 0; mi < 2; ++mi)
            af[mi] = frag_a(&As[(waveM * 32 + mi * 16) * SC_LD + ks], SC_LD, lane);
#pragma unroll
        for (int ni = 0; ni < 4; ++ni)
            bf[ni] = frag_b(&Bs[(waveN * 64 + ni * 16) * SC_LD + ks], SC_LD, lane);
#pragma unroll
        for (int mi = 0; mi < 2; ++mi)
#pragma unroll
            for (int ni = 0; ni < 4; ++ni)
                acc[mi][ni] = wmma32(af[mi], bf[ni], acc[mi][ni]);
    }

    const int ln = lane & 15, hi = lane >> 4;
    float* O = attn + (size_t)h * SEQ * SEQ;
#pragma unroll
    for (int mi = 0; mi < 2; ++mi)
#pragma unroll
        for (int ni = 0; ni < 4; ++ni)
#pragma unroll
            for (int g = 0; g < 8; ++g)
                O[(size_t)(tileM + waveM * 32 + mi * 16 + hi * 8 + g) * SEQ +
                  tileN + waveN * 64 + ni * 16 + ln] = acc[mi][ni][g] * 0.125f;
}

// ---------------- in-place row softmax over attn region --------------------
__global__ __launch_bounds__(256) void softmax_kernel(float* __restrict__ attn) {
    __shared__ float sm[256];
    const int tid = threadIdx.x;
    float* row = attn + (size_t)blockIdx.x * SEQ;
    float v[8];
#pragma unroll
    for (int i = 0; i < 8; ++i) v[i] = row[tid + i * 256];
    float m = v[0];
#pragma unroll
    for (int i = 1; i < 8; ++i) m = fmaxf(m, v[i]);
    sm[tid] = m; __syncthreads();
    for (int s = 128; s > 0; s >>= 1) {
        if (tid < s) sm[tid] = fmaxf(sm[tid], sm[tid + s]);
        __syncthreads();
    }
    const float rmax = sm[0]; __syncthreads();
    float lsum = 0.f;
#pragma unroll
    for (int i = 0; i < 8; ++i) { v[i] = __expf(v[i] - rmax); lsum += v[i]; }
    sm[tid] = lsum; __syncthreads();
    for (int s = 128; s > 0; s >>= 1) {
        if (tid < s) sm[tid] += sm[tid + s];
        __syncthreads();
    }
    const float inv = 1.0f / sm[0];
#pragma unroll
    for (int i = 0; i < 8; ++i) row[tid + i * 256] = v[i] * inv;
}

// ---------------- per-head ctx = attn(fp32->f16) @ v ; ctx f16 [m][h*64+d] -
__global__ __launch_bounds__(256) void attnv_kernel(const float* __restrict__ attn,
                                                    const _Float16* __restrict__ vT,
                                                    _Float16* __restrict__ ctx) {
    __shared__ _Float16 As[128 * SH_LD];
    __shared__ _Float16 Bs[64 * SH_LD];
    const int tid = threadIdx.x, lane = tid & 31, wave = tid >> 5;
    const int waveM = wave >> 1, waveN = wave & 1;           // wave: 32 x 32
    const int tileM = blockIdx.x * 128;
    const int h = blockIdx.z, kvh = h >> 2;
    const float*    A = attn + (size_t)h * SEQ * SEQ;
    const _Float16* B = vT + (size_t)(kvh * DK) * SEQ;       // rows d, ld = SEQ

    v8f acc[2][2] = {};
    for (int k0 = 0; k0 < SEQ; k0 += 32) {
#pragma unroll
        for (int c = tid; c < 1024; c += 256) {              // attn 128x32 fp32->f16
            int row = c >> 3, kc = (c & 7) * 4;
            float4 f = *(const float4*)&A[(size_t)(tileM + row) * SEQ + k0 + kc];
            v4h hv = {(_Float16)f.x, (_Float16)f.y, (_Float16)f.z, (_Float16)f.w};
            *(v4h*)&As[row * SH_LD + kc] = hv;
        }
        {
            int c = tid;                                     // vT 64x32 halves
            int row = c >> 2, kc = (c & 3) * 8;
            *(v8h*)&Bs[row * SH_LD + kc] =
                *(const v8h*)&B[(size_t)row * SEQ + k0 + kc];
        }
        __syncthreads();
        v16h af[2], bf[2];
#pragma unroll
        for (int mi = 0; mi < 2; ++mi)
            af[mi] = frag_a(&As[(waveM * 32 + mi * 16) * SH_LD], SH_LD, lane);
#pragma unroll
        for (int ni = 0; ni < 2; ++ni)
            bf[ni] = frag_b(&Bs[(waveN * 32 + ni * 16) * SH_LD], SH_LD, lane);
#pragma unroll
        for (int mi = 0; mi < 2; ++mi)
#pragma unroll
            for (int ni = 0; ni < 2; ++ni)
                acc[mi][ni] = wmma32(af[mi], bf[ni], acc[mi][ni]);
        __syncthreads();
    }

    const int ln = lane & 15, hi = lane >> 4;
#pragma unroll
    for (int mi = 0; mi < 2; ++mi)
#pragma unroll
        for (int ni = 0; ni < 2; ++ni)
#pragma unroll
            for (int g = 0; g < 8; ++g) {
                int m = tileM + waveM * 32 + mi * 16 + hi * 8 + g;
                int d = waveN * 32 + ni * 16 + ln;
                ctx[(size_t)m * DM + h * DK + d] = (_Float16)acc[mi][ni][g];
            }
}

// ---------------------------------------------------------------------------
extern "C" void kernel_launch(void* const* d_in, const int* in_sizes, int n_in,
                              void* d_out, int out_size, void* d_ws, size_t ws_size,
                              hipStream_t stream) {
    const float* x  = (const float*)d_in[0];
    const float* Wq = (const float*)d_in[1];
    const float* Wk = (const float*)d_in[2];
    const float* Wv = (const float*)d_in[3];
    const float* Wo = (const float*)d_in[4];

    float* out  = (float*)d_out;                         // [2048][2048]
    float* attn = out + (size_t)SEQ * DM;                // [32][2048][2048]

    const size_t MB = (size_t)1 << 20;
    char* w = (char*)d_ws;                               // 48 MB total
    _Float16* xh  = (_Float16*)(w + 0 * MB);             // [2048][2048]
    _Float16* WqT = (_Float16*)(w + 8 * MB);             // [2048][2048]
    _Float16* WkT = (_Float16*)(w + 16 * MB);            // [512][2048]
    _Float16* WvT = (_Float16*)(w + 18 * MB);            // [512][2048]
    _Float16* WoT = (_Float16*)(w + 20 * MB);            // [2048][2048]
    _Float16* qh  = (_Float16*)(w + 28 * MB);            // [2048][2048] rope'd
    _Float16* kh  = (_Float16*)(w + 36 * MB);            // [2048][512]  rope'd
    _Float16* vT  = (_Float16*)(w + 38 * MB);            // [512][2048]
    _Float16* ctx = (_Float16*)(w + 40 * MB);            // [2048][2048]

    dim3 blk(256);
    // convert / transpose weights + x to f16 (one-time)
    cvt_kernel<<<dim3(4096), blk, 0, stream>>>(x, xh, SEQ * DM);
    tcvt_kernel<<<dim3(64, 64), blk, 0, stream>>>(Wq, WqT, DM, DM);
    tcvt_kernel<<<dim3(16, 64), blk, 0, stream>>>(Wk, WkT, DM, NKVW);
    tcvt_kernel<<<dim3(16, 64), blk, 0, stream>>>(Wv, WvT, DM, NKVW);
    tcvt_kernel<<<dim3(64, 64), blk, 0, stream>>>(Wo, WoT, DM, DM);

    // projections (WMMA + async double-buffered staging)
    gemm_proj<1><<<dim3(16, 16), blk, 0, stream>>>(xh, WqT, qh, DM, DM);
    gemm_proj<1><<<dim3(4, 16),  blk, 0, stream>>>(xh, WkT, kh, DM, NKVW);
    gemm_proj<2><<<dim3(4, 16),  blk, 0, stream>>>(xh, WvT, vT, DM, SEQ);

    // attention scores -> d_out attn region (fp32, scaled)
    scores_kernel<<<dim3(16, 16, NHEADS), blk, 0, stream>>>(qh, kh, attn);
    // in-place softmax over 32*2048 rows
    softmax_kernel<<<dim3(NHEADS * SEQ), blk, 0, stream>>>(attn);
    // ctx = attn @ v (attn converted fp32->f16 during LDS staging)
    attnv_kernel<<<dim3(16, 1, NHEADS), blk, 0, stream>>>(attn, vT, ctx);
    // out = ctx @ Wo (fp32 store)
    gemm_proj<0><<<dim3(16, 16), blk, 0, stream>>>(ctx, WoT, out, DM, DM);

    (void)in_sizes; (void)n_in; (void)out_size; (void)ws_size;
}
